// PeriDynamicAttention_58737972740375
// MI455X (gfx1250) — matrix operand order repaired and checked by reading, not compiled, not run on hardware
//
#include <hip/hip_runtime.h>

#define TT    1024
#define NHEAD 16
#define BDIM  16
#define HSZ   64
#define DELTA 64
#define CEMB  1024

typedef __attribute__((ext_vector_type(16))) __bf16        v16bf;
typedef __attribute__((ext_vector_type(8)))  float         v8f;
typedef __attribute__((ext_vector_type(4)))  unsigned int  u32x4;

// ---------- bf16 helpers (bit-exact RNE, no native __bf16 arithmetic) ----------
__device__ __forceinline__ unsigned short f2bf_rne(float f) {
    union { float f; unsigned u; } v; v.f = f;
    unsigned u = v.u;
    unsigned rounding = 0x7FFFu + ((u >> 16) & 1u);
    u += rounding;
    return (unsigned short)(u >> 16);
}
__device__ __forceinline__ float bf2f(unsigned short h) {
    union { unsigned u; float f; } v; v.u = ((unsigned)h) << 16;
    return v.f;
}

// =====================================================================
// Split fp32 -> (hi, lo) bf16 pair, elementwise (row-major preserved).
// =====================================================================
__global__ __launch_bounds__(256)
void split_kernel(const float* __restrict__ in, unsigned short* __restrict__ hi,
                  unsigned short* __restrict__ lo, int n) {
    int i = blockIdx.x * 256 + threadIdx.x;
    if (i >= n) return;
    float f = in[i];
    unsigned short h = f2bf_rne(f);
    hi[i] = h;
    lo[i] = f2bf_rne(f - bf2f(h));
}

// =====================================================================
// Split + transpose: W (KxN) -> Wt_hi/lo (NxK) so WMMA B-fragments are
// contiguous 16B runs per lane.
// =====================================================================
__global__ __launch_bounds__(256)
void split_t_kernel(const float* __restrict__ in, unsigned short* __restrict__ hi,
                    unsigned short* __restrict__ lo, int K, int N) {
    int i = blockIdx.x * 256 + threadIdx.x;   // i = n*K + k
    if (i >= K * N) return;
    int n = i / K, k = i - n * K;
    float f = in[(size_t)k * N + n];
    unsigned short h = f2bf_rne(f);
    hi[i] = h;
    lo[i] = f2bf_rne(f - bf2f(h));
}

// =====================================================================
// GEMM  C(MxN) = A(MxK) @ W(KxN), pre-split bf16x3 operands.
// Block = 8 waves = 128x64 C macro-tile. The 64x32 B slab for each K
// step is staged into LDS with async global->LDS DMA, double-buffered
// on ASYNCcnt; waves read B fragments from LDS (ds_load_b128) and A
// fragments from global (b128), issuing 12 v_wmma per K=32 step.
// Requires M % 128 == 0, N % 64 == 0, K % 32 == 0.
// =====================================================================
__global__ __launch_bounds__(256)
void gemm_wmma(const unsigned short* __restrict__ Ahi, const unsigned short* __restrict__ Alo,
               const unsigned short* __restrict__ Bthi, const unsigned short* __restrict__ Btlo,
               float* __restrict__ C, int M, int K, int N) {
    // [buf][hi/lo][n_local * 32 + k_local]  -> 2*2*2048*2B = 16 KB
    __shared__ __align__(16) unsigned short sB[2][2][64 * 32];

    const int tid    = threadIdx.x;
    const int lane   = tid & 31;
    const int wid    = tid >> 5;
    const int tilesN = N >> 6;
    const int bm = (blockIdx.x / tilesN) << 7;   // 128-row macro tile
    const int tn = (blockIdx.x % tilesN) << 6;   // 64-col band
    const int tm = bm + (wid << 4);              // this wave's 16-row tile
    const int g   = lane >> 4;                   // half-wave K group
    const int l15 = lane & 15;

    const unsigned short* arow_h = Ahi + (size_t)(tm + l15) * K;
    const unsigned short* arow_l = Alo + (size_t)(tm + l15) * K;

    // ---- async stage of the 64x32 B slab (hi+lo = 8KB) for K step k0
    const int st_r    = tid >> 2;         // 0..63 : n_local row
    const int st_part = (tid & 3) * 8;    // ushort offset within row (0,8,16,24)
    auto stage = [&](int buf, int k0) {
        unsigned lds_h = (unsigned)(size_t)(&sB[buf][0][st_r * 32 + st_part]);
        unsigned lds_l = (unsigned)(size_t)(&sB[buf][1][st_r * 32 + st_part]);
        unsigned long long g_h = (unsigned long long)(const void*)
            (Bthi + (size_t)(tn + st_r) * K + k0 + st_part);
        unsigned long long g_l = (unsigned long long)(const void*)
            (Btlo + (size_t)(tn + st_r) * K + k0 + st_part);
        asm volatile("global_load_async_to_lds_b128 %0, %1, off"
                     :: "v"(lds_h), "v"(g_h) : "memory");
        asm volatile("global_load_async_to_lds_b128 %0, %1, off"
                     :: "v"(lds_l), "v"(g_l) : "memory");
    };

    v8f acc0 = {}, acc1 = {}, acc2 = {}, acc3 = {};
    stage(0, 0);
    for (int k0 = 0; k0 < K; k0 += 32) {
        const int cur = (k0 >> 5) & 1;
        if (k0 + 32 < K) {
            stage(cur ^ 1, k0 + 32);
            asm volatile("s_wait_asynccnt 0x2" ::: "memory");  // this step's slab landed
        } else {
            asm volatile("s_wait_asynccnt 0x0" ::: "memory");
        }
        __syncthreads();   // slab visible to all waves

        // A fragment: two 16B runs at k0+8g and k0+16+8g (hi and lo)
        union { v16bf v; u32x4 q[2]; } ah, al;
        const unsigned short* ap  = arow_h + k0 + 8 * g;
        const unsigned short* apl = arow_l + k0 + 8 * g;
        ah.q[0] = *(const u32x4*)(ap);
        ah.q[1] = *(const u32x4*)(ap + 16);
        al.q[0] = *(const u32x4*)(apl);
        al.q[1] = *(const u32x4*)(apl + 16);
        __builtin_prefetch(ap + 32, 0, 1);

#pragma unroll
        for (int nt = 0; nt < 4; ++nt) {
            const int nl = 16 * nt + l15;
            union { v16bf v; u32x4 q[2]; } bh, bl;
            bh.q[0] = *(const u32x4*)(&sB[cur][0][nl * 32 + 8 * g]);
            bh.q[1] = *(const u32x4*)(&sB[cur][0][nl * 32 + 8 * g + 16]);
            bl.q[0] = *(const u32x4*)(&sB[cur][1][nl * 32 + 8 * g]);
            bl.q[1] = *(const u32x4*)(&sB[cur][1][nl * 32 + 8 * g + 16]);
            v8f& acc = nt == 0 ? acc0 : nt == 1 ? acc1 : nt == 2 ? acc2 : acc3;
            acc = __builtin_amdgcn_wmma_f32_16x16x32_bf16(false, ah.v, false, bh.v,
                                                          (short)0, acc, false, false);
            acc = __builtin_amdgcn_wmma_f32_16x16x32_bf16(false, ah.v, false, bl.v,
                                                          (short)0, acc, false, false);
            acc = __builtin_amdgcn_wmma_f32_16x16x32_bf16(false, al.v, false, bh.v,
                                                          (short)0, acc, false, false);
        }
        __syncthreads();   // all waves done reading before next stage overwrites
    }
    // C/D layout: lanes 0-15 -> M = v, lanes 16-31 -> M = 8+v ; N = lane&15
#pragma unroll
    for (int nt = 0; nt < 4; ++nt) {
        const v8f& acc = nt == 0 ? acc0 : nt == 1 ? acc1 : nt == 2 ? acc2 : acc3;
#pragma unroll
        for (int v = 0; v < 8; ++v) {
            C[(size_t)(tm + 8 * g + v) * N + tn + 16 * nt + l15] = acc[v];
        }
    }
}

// =====================================================================
// posemb[j][o] = sum_d rel[j][d] * W_pos[d][o]    (64x16)
// =====================================================================
__global__ void posemb_kernel(const float* __restrict__ rel,
                              const float* __restrict__ wpos,
                              float* __restrict__ pe) {
    int i = blockIdx.x * blockDim.x + threadIdx.x;
    if (i >= DELTA * BDIM) return;
    int j = i >> 4, o = i & 15;
    float s = 0.f;
#pragma unroll
    for (int d = 0; d < 16; ++d) s = fmaf(rel[j * 16 + d], wpos[d * 16 + o], s);
    pe[i] = s;
}

// =====================================================================
// Windowed peridynamic attention. One wave per (b,h,t).
// Weight matrices stored TRANSPOSED in LDS so each output column's 16
// coefficients are contiguous (ds_load_b128-friendly, broadcast reads).
// =====================================================================
__global__ __launch_bounds__(256)
void window_attn(const float* __restrict__ disp, const float* __restrict__ val,
                 const float* __restrict__ Wstrain, const float* __restrict__ Wbond,
                 const float* __restrict__ Wdmg, const float* __restrict__ bdmg,
                 const float* __restrict__ Wdout, const float* __restrict__ bdout,
                 const float* __restrict__ pe, float* __restrict__ att) {
    __shared__ __align__(16) float sWs[256], sWd[256];  // [o][d] (transposed)
    __shared__ float sWb[16], sWdo[16], sbD[16];
    __shared__ __align__(16) float sPe[DELTA * BDIM];   // 4 KB

    // ---- CDNA5 async copy: 256 threads x 16B = 4KB posemb table -> LDS
    {
        unsigned ldsaddr = (unsigned)(size_t)(&sPe[0]) + threadIdx.x * 16u;
        unsigned long long ga = (unsigned long long)(const void*)pe +
                                (unsigned long long)threadIdx.x * 16ull;
        asm volatile("global_load_async_to_lds_b128 %0, %1, off"
                     :: "v"(ldsaddr), "v"(ga) : "memory");
    }
    {   // transpose 16x16 weights into LDS: s[o*16+d] = W[d*16+o]
        int i = threadIdx.x;
        if (i < 256) {
            int d = i >> 4, o = i & 15;
            sWs[o * 16 + d] = Wstrain[i];
            sWd[o * 16 + d] = Wdmg[i];
        }
    }
    if (threadIdx.x < 16) {
        sWb[threadIdx.x]  = Wbond[threadIdx.x];
        sWdo[threadIdx.x] = Wdout[threadIdx.x];
        sbD[threadIdx.x]  = bdmg[threadIdx.x];
    }
    asm volatile("s_wait_asynccnt 0x0" ::: "memory");
    __syncthreads();

    const float bout = bdout[0];
    const int lane = threadIdx.x & 31;
    const int wid  = threadIdx.x >> 5;
    const int idx  = blockIdx.x * 8 + wid;      // ((b*NHEAD)+h)*TT + t
    const int t  = idx & (TT - 1);
    const int bh = idx >> 10;
    const int h  = bh & (NHEAD - 1);
    const int b  = bh >> 4;

    const float* drow = disp + (size_t)(b * TT + t) * (NHEAD * BDIM) + h * BDIM;
    float dc[16];
#pragma unroll
    for (int d = 0; d < 16; ++d) dc[d] = drow[d];

    const float inv_sqrt2 = 0.70710678118654752f;
    float logit[2];
#pragma unroll
    for (int jj = 0; jj < 2; ++jj) {
        const int j  = lane + 32 * jj;
        const int tp = t - (DELTA - 1) + j;
        if (tp < 0) { logit[jj] = -__builtin_inff(); continue; }
        const float* dw = disp + (size_t)(b * TT + tp) * (NHEAD * BDIM) + h * BDIM;
        float st[16];
#pragma unroll
        for (int d = 0; d < 16; ++d) st[d] = dw[d] - dc[d];
        float bl = 0.f, dl = 0.f;
#pragma unroll
        for (int o = 0; o < 16; ++o) {
            float s1 = sPe[j * 16 + o];
            float s2 = sbD[o];
#pragma unroll
            for (int d = 0; d < 16; ++d) {
                s1 = fmaf(st[d], sWs[o * 16 + d], s1);   // contiguous in d
                s2 = fmaf(st[d], sWd[o * 16 + d], s2);
            }
            const float g1 = 0.5f * s1 * (1.f + erff(s1 * inv_sqrt2));
            const float g2 = 0.5f * s2 * (1.f + erff(s2 * inv_sqrt2));
            bl = fmaf(g1, sWb[o], bl);
            dl = fmaf(g2, sWdo[o], dl);
        }
        const float dmg = 1.f / (1.f + __expf(-(dl + bout)));
        logit[jj] = bl - 10.f * dmg;
    }

    // wave32 softmax over 64 logits (2 per lane)
    float mx = fmaxf(logit[0], logit[1]);
#pragma unroll
    for (int off = 16; off >= 1; off >>= 1) mx = fmaxf(mx, __shfl_xor(mx, off, 32));
    float e0 = __expf(logit[0] - mx);
    float e1 = __expf(logit[1] - mx);
    float s = e0 + e1;
#pragma unroll
    for (int off = 16; off >= 1; off >>= 1) s += __shfl_xor(s, off, 32);
    const float w0 = e0 / s, w1 = e1 / s;

    // out[e] = sum_j w_j * val[t-63+j][e]; lanes cover e = lane, lane+32
    float acc0 = 0.f, acc1 = 0.f;
    const float* vbase = val + (size_t)(b * TT) * CEMB + h * HSZ;
    for (int j = 0; j < DELTA; ++j) {
        const int tp = t - (DELTA - 1) + j;
        const float wj = __shfl((j < 32) ? w0 : w1, j & 31, 32);
        if (tp >= 0) {
            const float* vp = vbase + (size_t)tp * CEMB;
            acc0 = fmaf(wj, vp[lane], acc0);
            acc1 = fmaf(wj, vp[lane + 32], acc1);
        }
    }
    float* op = att + (size_t)(b * TT + t) * CEMB + h * HSZ;
    op[lane]      = acc0;
    op[lane + 32] = acc1;
}

// =====================================================================
extern "C" void kernel_launch(void* const* d_in, const int* in_sizes, int n_in,
                              void* d_out, int out_size, void* d_ws, size_t ws_size,
                              hipStream_t stream) {
    const float* x      = (const float*)d_in[0];
    const float* Wdisp  = (const float*)d_in[1];
    const float* Wval   = (const float*)d_in[2];
    const float* rel    = (const float*)d_in[3];
    const float* Wstr   = (const float*)d_in[4];
    const float* Wpos   = (const float*)d_in[5];
    const float* Wbond  = (const float*)d_in[6];
    const float* Wdmg   = (const float*)d_in[7];
    const float* bdmg   = (const float*)d_in[8];
    const float* Wdout  = (const float*)d_in[9];
    const float* bdout  = (const float*)d_in[10];
    const float* Wcproj = (const float*)d_in[11];
    float* out = (float*)d_out;

    const int M  = 2 * TT;          // 2048
    const int K  = CEMB;            // 1024
    const int ND = NHEAD * BDIM;    // 256

    // ---- workspace carve-up
    char* p = (char*)d_ws;
    auto alloc_f  = [&](size_t n) { float* r = (float*)p; p += n * sizeof(float); return r; };
    auto alloc_us = [&](size_t n) { unsigned short* r = (unsigned short*)p; p += n * sizeof(unsigned short); return r; };

    float* dispb = alloc_f((size_t)M * ND);
    float* valb  = alloc_f((size_t)M * CEMB);
    float* attb  = alloc_f((size_t)M * CEMB);
    float* peb   = alloc_f(DELTA * BDIM);
    unsigned short* xhi  = alloc_us((size_t)M * K);
    unsigned short* xlo  = alloc_us((size_t)M * K);
    unsigned short* athi = alloc_us((size_t)M * K);
    unsigned short* atlo = alloc_us((size_t)M * K);
    unsigned short* wdhi = alloc_us((size_t)ND * K);
    unsigned short* wdlo = alloc_us((size_t)ND * K);
    unsigned short* wvhi = alloc_us((size_t)CEMB * K);
    unsigned short* wvlo = alloc_us((size_t)CEMB * K);
    unsigned short* wphi = alloc_us((size_t)CEMB * K);
    unsigned short* wplo = alloc_us((size_t)CEMB * K);

    // ---- split inputs to bf16 hi/lo (weights also transposed)
    {
        int n = M * K;
        split_kernel<<<(n + 255) / 256, 256, 0, stream>>>(x, xhi, xlo, n);
    }
    split_t_kernel<<<(K * ND + 255) / 256, 256, 0, stream>>>(Wdisp, wdhi, wdlo, K, ND);
    split_t_kernel<<<(K * CEMB + 255) / 256, 256, 0, stream>>>(Wval, wvhi, wvlo, K, CEMB);
    split_t_kernel<<<(K * CEMB + 255) / 256, 256, 0, stream>>>(Wcproj, wphi, wplo, K, CEMB);

    // ---- disp = x @ W_disp ; val = x @ W_val   (grid = exact macro tiles)
    gemm_wmma<<<dim3((M / 128) * (ND / 64)), 256, 0, stream>>>(xhi, xlo, wdhi, wdlo, dispb, M, K, ND);
    gemm_wmma<<<dim3((M / 128) * (CEMB / 64)), 256, 0, stream>>>(xhi, xlo, wvhi, wvlo, valb, M, K, CEMB);
    posemb_kernel<<<(DELTA * BDIM + 255) / 256, 256, 0, stream>>>(rel, Wpos, peb);

    // ---- windowed attention
    {
        int nwaves = 2 * NHEAD * TT;
        window_attn<<<nwaves / 8, 256, 0, stream>>>(dispb, valb, Wstr, Wbond, Wdmg,
                                                    bdmg, Wdout, bdout, peb, attb);
    }

    // ---- out = att @ W_cproj
    {
        int n = M * K;
        split_kernel<<<(n + 255) / 256, 256, 0, stream>>>(attb, athi, atlo, n);
    }
    gemm_wmma<<<dim3((M / 128) * (CEMB / 64)), 256, 0, stream>>>(athi, atlo, wphi, wplo, out, M, K, CEMB);
}
